// AttentionMech_20435454395046
// MI455X (gfx1250) — matrix operand order repaired and checked by reading, not compiled, not run on hardware
//
#include <hip/hip_runtime.h>

// ---------------------------------------------------------------------------
// SAGAN-style attention for MI455X (gfx1250, wave32, WMMA).
//   x:[B,256,4096]  q,k:[B,32,4096]  v:[B,256,4096]
//   S[j,i] = sum_a k[a,j] q[a,i];  P = row-softmax_j over i of S
//   out[c,i] = sum_j v[c,j] P[j,i];  y = gamma*out + x
// Two-pass softmax (stats then fused recompute+GEMM), bf16 WMMA f32-accum.
// ---------------------------------------------------------------------------

#define B_ 4
#define C_ 256
#define A_ 32
#define N_ 4096
#define L2E 1.4426950408889634f

typedef __bf16 bf16_t;
typedef __attribute__((ext_vector_type(16))) __bf16 v16bf;
typedef __attribute__((ext_vector_type(8)))  __bf16 v8bf;
typedef __attribute__((ext_vector_type(8)))  float  v8f;

#if __has_builtin(__builtin_amdgcn_exp2f)
#define FAST_EXP2(x) __builtin_amdgcn_exp2f(x)
#else
#define FAST_EXP2(x) exp2f(x)
#endif

// A-matrix fragment (16x32 bf16, M=row per lane%16):
// lane half h: elements [0..7] = row[h*16 + t], [8..15] = row[h*16 + 8 + t]
__device__ __forceinline__ v16bf load_a_frag(const bf16_t* __restrict__ row, int half) {
  v8bf lo = *(const v8bf*)(row + half * 16);
  v8bf hi = *(const v8bf*)(row + half * 16 + 8);
  v16bf f;
#pragma unroll
  for (int t = 0; t < 8; ++t) { f[t] = lo[t]; f[t + 8] = hi[t]; }
  return f;
}

// B-matrix fragment (32x16 bf16, N=col per lane%16): K contiguous per half-wave
__device__ __forceinline__ v16bf load_b_frag(const bf16_t* __restrict__ row, int half) {
  return *(const v16bf*)(row + half * 16);
}

// ---------------- projection: q & k (VALU; 0.54 GFLOP) ----------------------
__global__ __launch_bounds__(256) void proj_qk_kernel(
    const float* __restrict__ x, const float* __restrict__ Wq,
    const float* __restrict__ bq, const float* __restrict__ Wk,
    const float* __restrict__ bk, bf16_t* __restrict__ Qt,
    bf16_t* __restrict__ Kt) {
  __shared__ float sWq[A_ * C_];
  __shared__ float sWk[A_ * C_];
  const int tid = threadIdx.x;
  for (int idx = tid; idx < A_ * C_; idx += 256) { sWq[idx] = Wq[idx]; sWk[idx] = Wk[idx]; }
  __syncthreads();

  const int b = blockIdx.y;
  const int n = blockIdx.x * 256 + tid;
  float qa[A_], ka[A_];
#pragma unroll
  for (int a = 0; a < A_; ++a) { qa[a] = 0.f; ka[a] = 0.f; }

  const float* xp = x + (size_t)b * C_ * N_ + n;
  for (int c = 0; c < C_; ++c) {
    const float xv = xp[(size_t)c * N_];
#pragma unroll
    for (int a = 0; a < A_; ++a) {
      qa[a] = fmaf(sWq[a * C_ + c], xv, qa[a]);
      ka[a] = fmaf(sWk[a * C_ + c], xv, ka[a]);
    }
  }
  bf16_t* qrow = Qt + ((size_t)b * N_ + n) * A_;
  bf16_t* krow = Kt + ((size_t)b * N_ + n) * A_;
#pragma unroll
  for (int ch = 0; ch < 4; ++ch) {
    v8bf pq, pk;
#pragma unroll
    for (int t = 0; t < 8; ++t) {
      pq[t] = (bf16_t)(qa[ch * 8 + t] + bq[ch * 8 + t]);
      pk[t] = (bf16_t)(ka[ch * 8 + t] + bk[ch * 8 + t]);
    }
    *(v8bf*)(qrow + ch * 8) = pq;
    *(v8bf*)(krow + ch * 8) = pk;
  }
}

// ---------------- projection: v (VALU; 2.1 GFLOP) ---------------------------
__global__ __launch_bounds__(256) void proj_v_kernel(
    const float* __restrict__ x, const float* __restrict__ Wv,
    const float* __restrict__ bv, bf16_t* __restrict__ V) {
  __shared__ float sW[32 * C_];
  const int tid = threadIdx.x;
  const int b = blockIdx.y;
  const int c0 = blockIdx.z * 32;
  for (int idx = tid; idx < 32 * C_; idx += 256) sW[idx] = Wv[(size_t)c0 * C_ + idx];
  __syncthreads();

  const int n = blockIdx.x * 256 + tid;
  float acc[32];
#pragma unroll
  for (int o = 0; o < 32; ++o) acc[o] = 0.f;

  const float* xp = x + (size_t)b * C_ * N_ + n;
  for (int c = 0; c < C_; ++c) {
    const float xv = xp[(size_t)c * N_];
#pragma unroll
    for (int o = 0; o < 32; ++o) acc[o] = fmaf(sW[o * C_ + c], xv, acc[o]);
  }
#pragma unroll
  for (int o = 0; o < 32; ++o)
    V[((size_t)b * C_ + c0 + o) * N_ + n] = (bf16_t)(acc[o] + bv[c0 + o]);
}

// -------- pass 1: per-j softmax stats over i (WMMA energy tiles) ------------
// D[m=i_loc, n=j_loc] = sum_a Q[a,i] K[a,j];  reduce over i per column j.
__global__ __launch_bounds__(128) void stats_kernel(
    const bf16_t* __restrict__ Qt, const bf16_t* __restrict__ Kt,
    float* __restrict__ statm, float* __restrict__ statz) {
  const int tid = threadIdx.x;
  const int wave = tid >> 5;
  const int lane = tid & 31;
  const int ln = lane & 15;
  const int half = lane >> 4;
  const int b = blockIdx.y;
  const int j = (blockIdx.x * 4 + wave) * 16 + ln;

  const v16bf bk_frag = load_b_frag(Kt + ((size_t)b * N_ + j) * A_, half);

  float mrun = -3.0e38f, zrun = 0.0f;
  for (int it = 0; it < N_ / 16; ++it) {
    const v16bf aq = load_a_frag(Qt + ((size_t)b * N_ + it * 16 + ln) * A_, half);
    v8f zc = (v8f)0.0f;
    v8f e = __builtin_amdgcn_wmma_f32_16x16x32_bf16(false, aq, false, bk_frag,
                                                    (short)0, zc, false, false);
    float tm = e[0];
#pragma unroll
    for (int t = 1; t < 8; ++t) tm = fmaxf(tm, e[t]);
    const float mnew = fmaxf(mrun, tm);
    const float scale = FAST_EXP2((mrun - mnew) * L2E);
    float s = 0.f;
#pragma unroll
    for (int t = 0; t < 8; ++t) s += FAST_EXP2((e[t] - mnew) * L2E);
    zrun = zrun * scale + s;
    mrun = mnew;
  }
  // combine lane l (rows 0..7) with lane l^16 (rows 8..15)
  const float mo = __shfl_xor(mrun, 16);
  const float zo = __shfl_xor(zrun, 16);
  const float mt = fmaxf(mrun, mo);
  const float zt = zrun * FAST_EXP2((mrun - mt) * L2E) + zo * FAST_EXP2((mo - mt) * L2E);
  if (half == 0) {
    statm[(size_t)b * N_ + j] = mt;
    statz[(size_t)b * N_ + j] = 1.0f / zt;
  }
}

// -------- pass 2: fused P recompute + out = V*P + epilogue ------------------
// 16 waves / WG; WG owns one 16-wide i tile; wave w owns c rows [16w,16w+16).
#define JC 512
__global__ __launch_bounds__(512) void out_kernel(
    const bf16_t* __restrict__ Qt, const bf16_t* __restrict__ Kt,
    const bf16_t* __restrict__ V, const float* __restrict__ statm,
    const float* __restrict__ statz, const float* __restrict__ x,
    const float* __restrict__ gamma, float* __restrict__ out) {
  __shared__ __attribute__((aligned(32))) bf16_t Pt[16][JC + 16];

  const int tid = threadIdx.x;
  const int w = tid >> 5;
  const int lane = tid & 31;
  const int ln = lane & 15;
  const int half = lane >> 4;
  const int b = blockIdx.y;
  const int i0 = blockIdx.x * 16;
  const int c0 = w * 16;

  // B-fragment of Q (cols i) for the S tiles -- invariant over the kernel
  const v16bf bq_frag = load_b_frag(Qt + ((size_t)b * N_ + i0 + ln) * A_, half);

  const float* smp = statm + (size_t)b * N_;
  const float* szp = statz + (size_t)b * N_;

  v8f acc = (v8f)0.0f;

  for (int jo = 0; jo < N_ / JC; ++jo) {
    const int jbase = jo * JC;

    // ---- phase 1: this wave computes P[jbase+w*32 .. +32, i tile] ----------
#pragma unroll
    for (int s = 0; s < 2; ++s) {
      const int jrow0 = jbase + w * 32 + s * 16;
      const v16bf ak = load_a_frag(Kt + ((size_t)b * N_ + jrow0 + ln) * A_, half);
      v8f zc = (v8f)0.0f;
      v8f sv = __builtin_amdgcn_wmma_f32_16x16x32_bf16(false, ak, false, bq_frag,
                                                       (short)0, zc, false, false);
      const int jg0 = jrow0 + half * 8;   // this lane's 8 j rows
      v8bf pk;
#pragma unroll
      for (int t = 0; t < 8; ++t) {
        const float mj = smp[jg0 + t];
        const float iz = szp[jg0 + t];
        pk[t] = (bf16_t)(FAST_EXP2((sv[t] - mj) * L2E) * iz);
      }
      *(v8bf*)&Pt[ln][w * 32 + s * 16 + half * 8] = pk;
    }
    __syncthreads();

    // ---- phase 2: acc += V[c tile, j chunk] * P[j chunk, i tile] -----------
#pragma unroll 4
    for (int ks = 0; ks < JC / 32; ++ks) {
      const int jj = ks * 32;
      const v16bf bp = *(const v16bf*)&Pt[ln][jj + half * 16];
      const v16bf av =
          load_a_frag(V + ((size_t)b * C_ + c0 + ln) * N_ + jbase + jj, half);
      acc = __builtin_amdgcn_wmma_f32_16x16x32_bf16(false, av, false, bp,
                                                    (short)0, acc, false, false);
    }
    __syncthreads();
  }

  // ---- epilogue: y = gamma*out + x  (D layout: col=lane%16, row=vgpr+8*half)
  const float g = gamma[0];
  const int i = i0 + ln;
#pragma unroll
  for (int t = 0; t < 8; ++t) {
    const int c = c0 + half * 8 + t;
    const size_t idx = ((size_t)b * C_ + c) * N_ + i;
    out[idx] = g * acc[t] + x[idx];
  }
}

// ---------------------------------------------------------------------------
extern "C" void kernel_launch(void* const* d_in, const int* in_sizes, int n_in,
                              void* d_out, int out_size, void* d_ws, size_t ws_size,
                              hipStream_t stream) {
  const float* x     = (const float*)d_in[0];
  const float* Wq    = (const float*)d_in[1];
  const float* bq    = (const float*)d_in[2];
  const float* Wk    = (const float*)d_in[3];
  const float* bk    = (const float*)d_in[4];
  const float* Wv    = (const float*)d_in[5];
  const float* bv    = (const float*)d_in[6];
  const float* gamma = (const float*)d_in[7];
  float* out = (float*)d_out;

  // workspace layout (~10.2 MB total)
  char* w = (char*)d_ws;
  bf16_t* Qt = (bf16_t*)(w);                        // [B][N][32] bf16 : 1 MB
  bf16_t* Kt = (bf16_t*)(w + (1u << 20));           // [B][N][32] bf16 : 1 MB
  bf16_t* V  = (bf16_t*)(w + (2u << 20));           // [B][C][N]  bf16 : 8 MB
  float* statm = (float*)(w + (10u << 20));         // [B][N] f32 : 64 KB
  float* statz = statm + (size_t)B_ * N_;           // [B][N] f32 : 64 KB

  proj_qk_kernel<<<dim3(N_ / 256, B_), 256, 0, stream>>>(x, Wq, bq, Wk, bk, Qt, Kt);
  proj_v_kernel<<<dim3(N_ / 256, B_, C_ / 32), 256, 0, stream>>>(x, Wv, bv, V);
  stats_kernel<<<dim3(N_ / 64, B_), 128, 0, stream>>>(Qt, Kt, statm, statz);
  out_kernel<<<dim3(N_ / 16, B_), 512, 0, stream>>>(Qt, Kt, V, statm, statz, x,
                                                    gamma, out);
}
